// MultiHeadSelfAttention_38439957300106
// MI455X (gfx1250) — matrix operand order repaired and checked by reading, not compiled, not run on hardware
//
#include <hip/hip_runtime.h>

// ---------------- types ----------------
typedef __attribute__((ext_vector_type(8)))  __bf16 v8bf;
typedef __attribute__((ext_vector_type(16))) __bf16 v16bf;
typedef __attribute__((ext_vector_type(8)))  float  v8f;
typedef __attribute__((ext_vector_type(4)))  unsigned int v4u;
typedef __attribute__((ext_vector_type(8)))  unsigned int v8u;

static __device__ inline v16bf cat8(v8bf lo, v8bf hi) {
  return __builtin_shufflevector(lo, hi, 0,1,2,3,4,5,6,7,8,9,10,11,12,13,14,15);
}

static __device__ inline v8f wmma_bf16(v16bf a, v16bf b, v8f c) {
  return __builtin_amdgcn_wmma_f32_16x16x32_bf16(false, a, false, b, (short)0, c, false, false);
}

static __device__ inline unsigned lds_off(const void* p) {
  // generic pointer to a __shared__ object: low 32 bits are the LDS byte offset
  return (unsigned)(unsigned long long)(uintptr_t)p;
}

// ---- Tensor Data Mover: 2D tile load Global -> LDS (D# per CDNA5 ISA) ----
// data_size = 2 bytes (bf16). padIC: pad_interval code (0=2dw,1=4,2=8,3=16,4=32,...);
// padAC: pad_amount code (0=1dw,1=2,...). All args must be wave-uniform; issue from ONE wave.
static __device__ inline void tdm_load_2d_bf16(
    unsigned lds_addr, const void* gaddr,
    unsigned tile_d0, unsigned tile_d1,
    unsigned tensor_d0, unsigned tensor_d1,
    unsigned stride0, unsigned padIC, unsigned padAC) {
  unsigned long long ga = (unsigned long long)(uintptr_t)gaddr;
  v4u g0;
  g0[0] = 1u;                                   // count=1, user descriptor
  g0[1] = lds_addr;                             // [63:32] lds_addr
  g0[2] = (unsigned)(ga & 0xFFFFFFFFu);         // [95:64] global_addr lo
  g0[3] = (unsigned)((ga >> 32) & 0x01FFFFFFu)  // [120:96] global_addr hi
        | (2u << 30);                           // [127:126] type = 2 (image)
  v8u g1;
  g1[0] = (1u << 16)                            // data_size = 2 bytes
        | (1u << 20)                            // pad_enable
        | ((padIC & 7u) << 22)                  // pad_interval
        | ((padAC & 0x7Fu) << 25);              // pad_amount
  g1[1] = (tensor_d0 & 0xFFFFu) << 16;          // [63:48] tensor_dim0 lo16 (abar=0)
  g1[2] = ((tensor_d0 >> 16) & 0xFFFFu)         // [79:64] tensor_dim0 hi16
        | ((tensor_d1 & 0xFFFFu) << 16);        // [95:80] tensor_dim1 lo16
  g1[3] = ((tensor_d1 >> 16) & 0xFFFFu)         // [111:96] tensor_dim1 hi16
        | ((tile_d0 & 0xFFFFu) << 16);          // [127:112] tile_dim0
  g1[4] = (tile_d1 & 0xFFFFu);                  // [143:128] tile_dim1 (tile_dim2=0)
  g1[5] = stride0;                              // [191:160] tensor_dim0_stride lo32
  g1[6] = 0u;                                   // stride hi16, dim1_stride lo16
  g1[7] = 0u;                                   // dim1_stride hi32
  asm volatile("tensor_load_to_lds %0, %1" :: "s"(g0), "s"(g1) : "memory");
}

#define D_MODEL 1024
#define N_HEADS 16
#define HEAD_DIM 64
#define SEQ_T 2048
#define NEG_INF (-1e30f)

// ---------------- f32 -> bf16 ----------------
__global__ void cvt_f32_bf16(const float* __restrict__ in, __bf16* __restrict__ out, int n) {
  int i = blockIdx.x * blockDim.x + threadIdx.x;
  if (i < n) out[i] = (__bf16)in[i];
}

// ---------------- tiled bf16 GEMM:  out = A[M,K] * W[N,K]^T + bias ----------------
// block = 256 threads (8 waves). BM=128, BN=128, BK=32. TDM-staged, double-buffered.
// wave w: wr=w/2 -> 32-row strip, wc=w%2 -> 64-col strip. 2x4 WMMA tiles per wave.
#define GLDA 40   // padded LDS row stride in elements (64B data + 16B TDM pad = 80B)
__global__ __launch_bounds__(256) void gemm_bf16_wmma(
    const __bf16* __restrict__ A,     // [M,K] row-major
    const __bf16* __restrict__ W,     // [N,K] row-major  (== column-major B)
    const float*  __restrict__ bias,  // [N]
    __bf16* __restrict__ outSplit,    // [B,H,T,Hd] bf16 when split==1
    float*  __restrict__ outF32,      // [M,N] f32 when split==0
    int M, int N, int K, int split) {
  __shared__ __bf16 As[2][128 * GLDA];
  __shared__ __bf16 Bs[2][128 * GLDA];

  const int tid  = threadIdx.x;
  const int wave = tid >> 5;
  const int lane = tid & 31;
  const int half = lane >> 4;   // 0: lanes 0-15, 1: lanes 16-31
  const int l16  = lane & 15;
  const int wr   = wave >> 1;   // 0..3
  const int wc   = wave & 1;    // 0..1
  const int m0   = blockIdx.x * 128;
  const int n0   = blockIdx.y * 128;

  v8f acc[2][4];
#pragma unroll
  for (int i = 0; i < 2; ++i)
#pragma unroll
    for (int j = 0; j < 4; ++j) acc[i][j] = (v8f)(0.0f);

  const int kIters = K / 32;

  // preload slab 0 via TDM (pad 16 dwords -> +4 dwords => 80B LDS row stride)
  if (wave == 0) {
    tdm_load_2d_bf16(lds_off(&As[0][0]), A + (size_t)m0 * K, 32u, 128u,
                     (unsigned)K, (unsigned)M, (unsigned)K, 3u, 3u);
    tdm_load_2d_bf16(lds_off(&Bs[0][0]), W + (size_t)n0 * K, 32u, 128u,
                     (unsigned)K, (unsigned)N, (unsigned)K, 3u, 3u);
    __builtin_amdgcn_s_wait_tensorcnt(0);
  }
  __syncthreads();

  for (int it = 0; it < kIters; ++it) {
    const int cur = it & 1;
    // issue next slab's DMA into the other buffer (its readers fenced last barrier)
    if (wave == 0 && it + 1 < kIters) {
      const int k0n = (it + 1) * 32;
      tdm_load_2d_bf16(lds_off(&As[cur ^ 1][0]), A + (size_t)m0 * K + k0n, 32u, 128u,
                       (unsigned)K, (unsigned)M, (unsigned)K, 3u, 3u);
      tdm_load_2d_bf16(lds_off(&Bs[cur ^ 1][0]), W + (size_t)n0 * K + k0n, 32u, 128u,
                       (unsigned)K, (unsigned)N, (unsigned)K, 3u, 3u);
    }

    // ---- build fragments from current buffer ----
    const __bf16* Ac = &As[cur][0];
    const __bf16* Bc = &Bs[cur][0];
    v16bf afrag[2], bfrag[4];
#pragma unroll
    for (int i = 0; i < 2; ++i) {
      // A 16x32 layout: lanes<16 rows 0-15 K{0..7,16..23}; lanes>=16 K{8..15,24..31}
      int r  = wr * 32 + i * 16 + l16;
      int kb = half * 8;
      v8bf lo = *(const v8bf*)(&Ac[r * GLDA + kb]);
      v8bf hi = *(const v8bf*)(&Ac[r * GLDA + kb + 16]);
      afrag[i] = cat8(lo, hi);
    }
#pragma unroll
    for (int j = 0; j < 4; ++j) {
      // B 32x16 layout: lane col n=l16; lanes<16 K{0..7,16..23}; lanes>=16 K{8..15,24..31}
      int n  = wc * 64 + j * 16 + l16;
      int kb = half * 8;
      v8bf lo = *(const v8bf*)(&Bc[n * GLDA + kb]);
      v8bf hi = *(const v8bf*)(&Bc[n * GLDA + kb + 16]);
      bfrag[j] = cat8(lo, hi);
    }
#pragma unroll
    for (int i = 0; i < 2; ++i)
#pragma unroll
      for (int j = 0; j < 4; ++j)
        acc[i][j] = wmma_bf16(afrag[i], bfrag[j], acc[i][j]);

    if (wave == 0) __builtin_amdgcn_s_wait_tensorcnt(0);
    __syncthreads();
  }

  // ---- epilogue: C layout row = vgpr + 8*half, col = l16 ----
#pragma unroll
  for (int i = 0; i < 2; ++i) {
#pragma unroll
    for (int j = 0; j < 4; ++j) {
      int gn = n0 + wc * 64 + j * 16 + l16;
      float bv = bias[gn];
#pragma unroll
      for (int r = 0; r < 8; ++r) {
        int gm = m0 + wr * 32 + i * 16 + r + half * 8;
        float v = acc[i][j][r] + bv;
        if (split) {
          int bb = gm >> 11;            // / SEQ_T
          int t  = gm & (SEQ_T - 1);
          int h  = gn >> 6;             // / HEAD_DIM
          int d  = gn & (HEAD_DIM - 1);
          outSplit[(((size_t)(bb * N_HEADS + h) * SEQ_T + t) << 6) + d] = (__bf16)v;
        } else {
          outF32[(size_t)gm * N + gn] = v;
        }
      }
    }
  }
}

// ---------------- flash attention (bf16 WMMA, f32 online softmax) ----------------
// grid: (T/128, B*H). block 256 = 8 waves; wave w owns 16 query rows.
#define KLD 72    // Ks row stride (128B data + 16B TDM pad = 144B)
#define VLD 136   // Vt / P row stride (128 + 8 pad elements)
__global__ __launch_bounds__(256) void attn_wmma(
    const __bf16* __restrict__ Qh,   // [B,H,T,64]
    const __bf16* __restrict__ Kh,
    const __bf16* __restrict__ Vh,
    const int*    __restrict__ amask,  // [B,T]
    __bf16* __restrict__ ctx) {        // [B,T,D] merged heads
  __shared__ __bf16 Ks[128 * KLD];      // K tile, row-major [s][d] (TDM-staged)
  __shared__ __bf16 Vt[64 * VLD];       // V tile transposed [d][s]
  __shared__ __bf16 Pws[8][16 * VLD];   // per-wave P staging [qrow][s]
  __shared__ float  kbias[128];

  const int bh  = blockIdx.y;            // 0..31
  const int bb  = bh >> 4;
  const int h   = bh & 15;
  const int qb  = blockIdx.x;            // 0..15
  const int tid = threadIdx.x;
  const int wave = tid >> 5;
  const int lane = tid & 31;
  const int half = lane >> 4;
  const int l16  = lane & 15;

  const __bf16* Qbase = Qh + (size_t)bh * SEQ_T * HEAD_DIM;
  const __bf16* Kbase = Kh + (size_t)bh * SEQ_T * HEAD_DIM;
  const __bf16* Vbase = Vh + (size_t)bh * SEQ_T * HEAD_DIM;

  const int q0 = qb * 128 + wave * 16;   // this wave's first query row

  // q fragments (A layout), K=64 -> 2 frags, loaded straight from global
  v16bf qa[2];
#pragma unroll
  for (int ka = 0; ka < 2; ++ka) {
    const __bf16* src = Qbase + (size_t)(q0 + l16) * HEAD_DIM + ka * 32 + half * 8;
    v8bf lo = *(const v8bf*)(src);
    v8bf hi = *(const v8bf*)(src + 16);
    qa[ka] = cat8(lo, hi);
  }

  float mrow[8], lrow[8];
  v8f cacc[4];
#pragma unroll
  for (int r = 0; r < 8; ++r) { mrow[r] = NEG_INF; lrow[r] = 0.0f; }
#pragma unroll
  for (int j = 0; j < 4; ++j) cacc[j] = (v8f)(0.0f);

  for (int sb = 0; sb <= qb; ++sb) {
    const int s0 = sb * 128;
    __syncthreads();                       // prev tile fully consumed
    // ---- stage K via TDM (pad 32 dwords -> +4 dwords => 144B row stride) ----
    if (wave == 0) {
      tdm_load_2d_bf16(lds_off(&Ks[0]), Kbase + (size_t)s0 * HEAD_DIM, 64u, 128u,
                       64u, (unsigned)SEQ_T, 64u, 4u, 3u);
    }
    // ---- stage V transposed (all threads) ----
    {
      int s  = tid >> 1;
      int c0 = (tid & 1) * 32;
      const __bf16* vsrc = Vbase + (size_t)(s0 + s) * HEAD_DIM + c0;
      v8bf vv[4];
#pragma unroll
      for (int u = 0; u < 4; ++u) vv[u] = *(const v8bf*)(vsrc + u * 8);
#pragma unroll
      for (int u = 0; u < 4; ++u)
#pragma unroll
        for (int e = 0; e < 8; ++e)
          Vt[(c0 + u * 8 + e) * VLD + s] = vv[u][e];
    }
    if (tid < 128) kbias[tid] = amask[bb * SEQ_T + s0 + tid] ? 0.0f : NEG_INF;
    if (wave == 0) __builtin_amdgcn_s_wait_tensorcnt(0);
    __syncthreads();

    // ---- scores: 16 x 128, QK^T via 8 N-tiles x 2 K-chunks ----
    v8f sacc[8];
#pragma unroll
    for (int j = 0; j < 8; ++j) {
      sacc[j] = (v8f)(0.0f);
#pragma unroll
      for (int kd = 0; kd < 2; ++kd) {
        // B frag: col = key s (contig d in Ks row), K chunk kd
        const __bf16* p = &Ks[(j * 16 + l16) * KLD + kd * 32 + half * 8];
        v8bf lo = *(const v8bf*)(p);
        v8bf hi = *(const v8bf*)(p + 16);
        sacc[j] = wmma_bf16(qa[kd], cat8(lo, hi), sacc[j]);
      }
    }

    // ---- scale, mask, online softmax ----
#pragma unroll
    for (int r = 0; r < 8; ++r) {
      const int qrow = q0 + r + half * 8;
      float mx = NEG_INF;
#pragma unroll
      for (int j = 0; j < 8; ++j) {
        int sl = j * 16 + l16;
        float sc = sacc[j][r] * 0.125f + kbias[sl];
        if (s0 + sl > qrow) sc = NEG_INF;     // causal
        sacc[j][r] = sc;
        mx = fmaxf(mx, sc);
      }
#pragma unroll
      for (int off = 1; off < 16; off <<= 1)
        mx = fmaxf(mx, __shfl_xor(mx, off, 32));
      float mnew = fmaxf(mrow[r], mx);
      float alpha = __expf(mrow[r] - mnew);
      float rs = 0.0f;
#pragma unroll
      for (int j = 0; j < 8; ++j) {
        float e = __expf(sacc[j][r] - mnew);
        sacc[j][r] = e;
        rs += e;
      }
#pragma unroll
      for (int off = 1; off < 16; off <<= 1)
        rs += __shfl_xor(rs, off, 32);
      lrow[r] = lrow[r] * alpha + rs;
      mrow[r] = mnew;
#pragma unroll
      for (int jd = 0; jd < 4; ++jd) cacc[jd][r] *= alpha;
    }

    // ---- stage P (C layout -> LDS row-major) then reload as A frags ----
    __bf16* P = Pws[wave];
#pragma unroll
    for (int r = 0; r < 8; ++r)
#pragma unroll
      for (int j = 0; j < 8; ++j)
        P[(r + half * 8) * VLD + j * 16 + l16] = (__bf16)sacc[j][r];

#pragma unroll
    for (int ka2 = 0; ka2 < 4; ++ka2) {
      const __bf16* pr = &P[l16 * VLD + ka2 * 32 + half * 8];
      v8bf lo = *(const v8bf*)(pr);
      v8bf hi = *(const v8bf*)(pr + 16);
      v16bf pa = cat8(lo, hi);
#pragma unroll
      for (int jd = 0; jd < 4; ++jd) {
        // B frag from Vt: col d, K=s in table order (half*8, +16)
        const __bf16* vb = &Vt[(jd * 16 + l16) * VLD + ka2 * 32 + half * 8];
        v8bf vlo = *(const v8bf*)(vb);
        v8bf vhi = *(const v8bf*)(vb + 16);
        cacc[jd] = wmma_bf16(pa, cat8(vlo, vhi), cacc[jd]);
      }
    }
  }

  // ---- finalize: ctx[b, t, h*64 + d] = cacc / l ----
#pragma unroll
  for (int jd = 0; jd < 4; ++jd) {
#pragma unroll
    for (int r = 0; r < 8; ++r) {
      int t = q0 + r + half * 8;
      int d = jd * 16 + l16;
      float v = cacc[jd][r] / lrow[r];
      ctx[((size_t)(bb * SEQ_T + t)) * D_MODEL + h * HEAD_DIM + d] = (__bf16)v;
    }
  }
}

// ---------------- launcher ----------------
extern "C" void kernel_launch(void* const* d_in, const int* in_sizes, int n_in,
                              void* d_out, int out_size, void* d_ws, size_t ws_size,
                              hipStream_t stream) {
  const float* x     = (const float*)d_in[0];
  const int*   amask = (const int*)  d_in[1];
  const float* Wq    = (const float*)d_in[2];
  const float* bq    = (const float*)d_in[3];
  const float* Wk    = (const float*)d_in[4];
  const float* bk    = (const float*)d_in[5];
  const float* Wv    = (const float*)d_in[6];
  const float* bv    = (const float*)d_in[7];
  const float* Wo    = (const float*)d_in[8];
  const float* bo    = (const float*)d_in[9];

  const int B = 2, T = SEQ_T, D = D_MODEL;
  const int M = B * T;            // 4096
  const size_t nx = (size_t)M * D;
  const size_t nw = (size_t)D * D;
  const size_t nh = (size_t)B * N_HEADS * T * HEAD_DIM;  // == nx

  char* ws = (char*)d_ws;
  size_t off = 0;
  auto carve = [&](size_t bytes) -> void* {
    void* p = ws + off;
    off += (bytes + 255) & ~(size_t)255;
    return p;
  };
  __bf16* Xb  = (__bf16*)carve(nx * 2);
  __bf16* Wqb = (__bf16*)carve(nw * 2);
  __bf16* Wkb = (__bf16*)carve(nw * 2);
  __bf16* Wvb = (__bf16*)carve(nw * 2);
  __bf16* Wob = (__bf16*)carve(nw * 2);
  __bf16* Qhp = (__bf16*)carve(nh * 2);
  __bf16* Khp = (__bf16*)carve(nh * 2);
  __bf16* Vhp = (__bf16*)carve(nh * 2);
  __bf16* Ctx = (__bf16*)carve(nx * 2);
  (void)ws_size;

  // 1) convert inputs to bf16
  cvt_f32_bf16<<<(int)((nx + 255) / 256), 256, 0, stream>>>(x,  Xb,  (int)nx);
  cvt_f32_bf16<<<(int)((nw + 255) / 256), 256, 0, stream>>>(Wq, Wqb, (int)nw);
  cvt_f32_bf16<<<(int)((nw + 255) / 256), 256, 0, stream>>>(Wk, Wkb, (int)nw);
  cvt_f32_bf16<<<(int)((nw + 255) / 256), 256, 0, stream>>>(Wv, Wvb, (int)nw);
  cvt_f32_bf16<<<(int)((nw + 255) / 256), 256, 0, stream>>>(Wo, Wob, (int)nw);

  // 2) Q/K/V projections -> head-split bf16
  dim3 gg(M / 128, D / 128);   // 32 x 8
  gemm_bf16_wmma<<<gg, 256, 0, stream>>>(Xb, Wqb, bq, Qhp, nullptr, M, D, D, 1);
  gemm_bf16_wmma<<<gg, 256, 0, stream>>>(Xb, Wkb, bk, Khp, nullptr, M, D, D, 1);
  gemm_bf16_wmma<<<gg, 256, 0, stream>>>(Xb, Wvb, bv, Vhp, nullptr, M, D, D, 1);

  // 3) causal flash attention -> merged-head bf16 context
  attn_wmma<<<dim3(T / 128, B * N_HEADS), 256, 0, stream>>>(Qhp, Khp, Vhp, amask, Ctx);

  // 4) output projection -> f32
  gemm_bf16_wmma<<<gg, 256, 0, stream>>>(Ctx, Wob, bo, nullptr, (float*)d_out, M, D, D, 0);
}
